// MultiheadedSelfAttention_59579786330250
// MI455X (gfx1250) — compile-verified, matrix operand order checked by reading, not run on hardware
//
#include <hip/hip_runtime.h>

typedef __attribute__((ext_vector_type(16))) __bf16 v16bf;
typedef __attribute__((ext_vector_type(8)))  __bf16 v8bf;
typedef __attribute__((ext_vector_type(8)))  float  v8f;

#define NHEAD 12
#define EMBED 768
#define HD    64
#define BATCH 8
#define SEQ   1025
#define MTOT  (BATCH*SEQ)   /* 8200 */
#define MPAD  8224          /* MTOT rounded to 32 so 32-row A slabs need no guard */
#define SPAD  1056          /* seq padded to 33*32: unguarded b128 loads + clean chunks */

// Build a v16 bf16 fragment from two contiguous 16-byte chunks.
// ISA 7.12.2 (wave32, 16-bit A/B 16x32): lane half selects kbase {0,8};
// elements 0..7 <- K = kbase+i, elements 8..15 <- K = 16+kbase+(i-8).
static __device__ __forceinline__ v16bf load_frag(const __bf16* base, int lo, int hi) {
  v8bf a = *(const v8bf*)(base + lo);
  v8bf b = *(const v8bf*)(base + hi);
  return __builtin_shufflevector(a, b, 0,1,2,3,4,5,6,7,8,9,10,11,12,13,14,15);
}

static __device__ __forceinline__ v8f wmma_bf16(v16bf a, v16bf b, v8f c) {
  return __builtin_amdgcn_wmma_f32_16x16x32_bf16(false, a, false, b, (short)0, c, false, false);
}

// Fragment set for one 32(M)x64(N)x32(K) step: 2 A fragments + 4 B fragments.
struct GemmFrags { v16bf a0, a1, b0, b1, b2, b3; };

static __device__ __forceinline__ void load_step(GemmFrags& f, const __bf16* xr0,
                                                 const __bf16* xr1, const __bf16* w0,
                                                 int kk, int kb) {
  f.a0 = load_frag(xr0, kk + kb, kk + 16 + kb);
  f.a1 = load_frag(xr1, kk + kb, kk + 16 + kb);
  f.b0 = load_frag(w0,                             kk + kb, kk + 16 + kb);
  f.b1 = load_frag(w0 + (size_t)16 * EMBED,        kk + kb, kk + 16 + kb);
  f.b2 = load_frag(w0 + (size_t)32 * EMBED,        kk + kb, kk + 16 + kb);
  f.b3 = load_frag(w0 + (size_t)48 * EMBED,        kk + kb, kk + 16 + kb);
}

static __device__ __forceinline__ void mma_step(const GemmFrags& f, v8f acc[2][4]) {
  acc[0][0] = wmma_bf16(f.a0, f.b0, acc[0][0]);
  acc[1][0] = wmma_bf16(f.a1, f.b0, acc[1][0]);
  acc[0][1] = wmma_bf16(f.a0, f.b1, acc[0][1]);
  acc[1][1] = wmma_bf16(f.a1, f.b1, acc[1][1]);
  acc[0][2] = wmma_bf16(f.a0, f.b2, acc[0][2]);
  acc[1][2] = wmma_bf16(f.a1, f.b2, acc[1][2]);
  acc[0][3] = wmma_bf16(f.a0, f.b3, acc[0][3]);
  acc[1][3] = wmma_bf16(f.a1, f.b3, acc[1][3]);
}

// Explicit ping-pong over the K dimension (EMBED/32 = 24 steps, even):
// loads land directly in the consumed register set -> no rotation copies.
static __device__ __forceinline__ void gemm_loop(v8f acc[2][4], const __bf16* xr0,
                                                 const __bf16* xr1, const __bf16* w0,
                                                 int kb) {
  GemmFrags f0, f1;
  load_step(f0, xr0, xr1, w0, 0, kb);
  for (int kk = 32; kk + 64 <= EMBED; kk += 64) {
    load_step(f1, xr0, xr1, w0, kk, kb);
    mma_step(f0, acc);
    load_step(f0, xr0, xr1, w0, kk + 32, kb);
    mma_step(f1, acc);
  }
  load_step(f1, xr0, xr1, w0, EMBED - 32, kb);
  mma_step(f0, acc);
  mma_step(f1, acc);
}

__global__ void cvt_bf16(const float* __restrict__ src, __bf16* __restrict__ dst, int n) {
  int i = blockIdx.x * blockDim.x + threadIdx.x;
  if (i < n) dst[i] = (__bf16)src[i];
}

// y = x @ W^T + b for Q/K/V (z selects weight). One wave per 32x64 output
// slab: 4 B fragments shared across 2 A fragments (8 WMMAs / 12 b128 loads
// per K-step), ping-pong pipelined. __launch_bounds__(128,1): full VGPR
// budget — these waves self-hide latency, no spills wanted.
__global__ void __launch_bounds__(128, 1)
qkv_gemm(const __bf16* __restrict__ xb,
         const __bf16* __restrict__ Wqb, const __bf16* __restrict__ Wkb,
         const __bf16* __restrict__ Wvb,
         const float* __restrict__ bq, const float* __restrict__ bk,
         const float* __restrict__ bv,
         __bf16* __restrict__ qo, __bf16* __restrict__ ko,
         __bf16* __restrict__ vto) {
  const int lane = threadIdx.x;
  const int ln   = lane & 15;
  const int half = lane >> 4;
  const int kb   = half * 8;
  const int mt   = blockIdx.x;
  const int n64  = blockIdx.y * blockDim.y + threadIdx.y;   // 0..11 == head index
  const int z    = blockIdx.z;
  const __bf16* W    = (z == 0) ? Wqb : (z == 1) ? Wkb : Wvb;
  const float*  bias = (z == 0) ? bq  : (z == 1) ? bk  : bv;

  const __bf16* xr0 = xb + (size_t)(mt * 32 + ln) * EMBED;
  const __bf16* xr1 = xr0 + (size_t)16 * EMBED;
  const __bf16* w0  = W + (size_t)(n64 * 64 + ln) * EMBED;

  v8f acc[2][4] = {};
  gemm_loop(acc, xr0, xr1, w0, kb);

  const int h = n64;                       // 64-col groups align with heads
#pragma unroll
  for (int i = 0; i < 2; ++i) {
#pragma unroll
    for (int j = 0; j < 4; ++j) {
      const int hd = j * 16 + ln;
      const float bv_ = bias[n64 * 64 + hd];
#pragma unroll
      for (int r = 0; r < 8; ++r) {        // C-layout: row = r + 8*half
        int mr = mt * 32 + i * 16 + r + 8 * half;
        if (mr >= MTOT) break;
        int bb = mr / SEQ;
        int s  = mr - bb * SEQ;
        float y = acc[i][j][r] + bv_;
        size_t bh = (size_t)(bb * NHEAD + h);
        if (z == 0)      qo [(bh * SPAD + s) * HD + hd]  = (__bf16)(y * 0.125f);
        else if (z == 1) ko [(bh * SPAD + s) * HD + hd]  = (__bf16)y;
        else             vto[(bh * HD + hd) * SPAD + s]  = (__bf16)y;
      }
    }
  }
}

// Flash-style attention: S^T = K * Q^T so exp(S^T) lands directly in the
// P A-fragment layout; online softmax; O += P * V per 32-key chunk.
// V loads hoisted above softmax VALU; masking only in the final chunk.
__global__ void __launch_bounds__(128, 1)
attn_kernel(const __bf16* __restrict__ qd, const __bf16* __restrict__ kd,
            const __bf16* __restrict__ vtd, __bf16* __restrict__ ctx) {
  const int lane = threadIdx.x;
  const int ln   = lane & 15;
  const int half = lane >> 4;
  const int kb   = half * 8;
  const int qt   = blockIdx.x * blockDim.y + threadIdx.y;
  if (qt > (SEQ - 1) / 16) return;         // wave-uniform guard
  const int h = blockIdx.y, b = blockIdx.z;

  const __bf16* Qb = qd  + ((size_t)(b * NHEAD + h) * SPAD) * HD;
  const __bf16* Kb = kd  + ((size_t)(b * NHEAD + h) * SPAD) * HD;
  const __bf16* Vb = vtd + ((size_t)(b * NHEAD + h) * HD) * SPAD;

  // Q^T as B-fragments (lane = query column), two K=32 chunks covering hd 0..63
  const __bf16* qr = Qb + (size_t)(qt * 16 + ln) * HD;
  v16bf qb0 = load_frag(qr, kb, 16 + kb);
  v16bf qb1 = load_frag(qr, 32 + kb, 48 + kb);

  float Mr = -1e30f, Lr = 0.0f;
  v8f o[4] = {};                           // O cols hd = j*16 + ln, rows = queries
  for (int k0 = 0; k0 < SPAD; k0 += 32) {
    const __bf16* kr0 = Kb + (size_t)(k0 + ln) * HD;
    const __bf16* kr1 = kr0 + 16 * HD;
    if (k0 + 32 < SPAD) __builtin_prefetch(kr0 + 32 * HD, 0, 3);
    v8f sT0 = {}, sT1 = {};
    sT0 = wmma_bf16(load_frag(kr0, kb, 16 + kb), qb0, sT0);
    sT0 = wmma_bf16(load_frag(kr0, 32 + kb, 48 + kb), qb1, sT0);
    sT1 = wmma_bf16(load_frag(kr1, kb, 16 + kb), qb0, sT1);
    sT1 = wmma_bf16(load_frag(kr1, 32 + kb, 48 + kb), qb1, sT1);

    // hoist V fragment loads: softmax VALU below hides their latency
    v16bf vb[4];
#pragma unroll
    for (int j = 0; j < 4; ++j)
      vb[j] = load_frag(Vb + (size_t)(j * 16 + ln) * SPAD, k0 + kb, k0 + 16 + kb);

    if (k0 + 32 > SEQ) {                   // only the final chunk has padded keys
#pragma unroll
      for (int r = 0; r < 8; ++r) {
        int kg = k0 + r + 8 * half;
        if (kg >= SEQ)      sT0[r] = -1e30f;
        if (kg + 16 >= SEQ) sT1[r] = -1e30f;
      }
    }
    float cmax = -1e30f;
#pragma unroll
    for (int r = 0; r < 8; ++r)
      cmax = fmaxf(cmax, fmaxf(sT0[r], sT1[r]));
    cmax = fmaxf(cmax, __shfl_xor(cmax, 16, 32));
    float newM  = fmaxf(Mr, cmax);
    float alpha = __expf(Mr - newM);
    float psum  = 0.0f;
    v16bf pa;                               // P in A-fragment layout, no shuffles
#pragma unroll
    for (int r = 0; r < 8; ++r) {
      float p0 = __expf(sT0[r] - newM);
      float p1 = __expf(sT1[r] - newM);
      psum += p0 + p1;
      pa[r]     = (__bf16)p0;
      pa[r + 8] = (__bf16)p1;
    }
    psum += __shfl_xor(psum, 16, 32);
    Lr = Lr * alpha + psum;
    Mr = newM;

    float arow[8];
#pragma unroll
    for (int r = 0; r < 8; ++r) arow[r] = __shfl(alpha, r + 8 * half, 32);

#pragma unroll
    for (int j = 0; j < 4; ++j) {           // O[:, j*16..] += P * V_chunk
      v8f oc;
#pragma unroll
      for (int r = 0; r < 8; ++r) oc[r] = o[j][r] * arow[r];
      o[j] = wmma_bf16(pa, vb[j], oc);
    }
  }

  float lrow[8];
#pragma unroll
  for (int r = 0; r < 8; ++r) lrow[r] = 1.0f / __shfl(Lr, r + 8 * half, 32);
#pragma unroll
  for (int j = 0; j < 4; ++j) {
    int hd = j * 16 + ln;
#pragma unroll
    for (int r = 0; r < 8; ++r) {
      int qg = qt * 16 + r + 8 * half;
      if (qg < SEQ)
        ctx[((size_t)b * SEQ + qg) * EMBED + h * HD + hd] = (__bf16)(o[j][r] * lrow[r]);
    }
  }
}

// out = ctx @ Wp^T + bp (f32 result), 32x64 per wave, ping-pong pipelined
__global__ void __launch_bounds__(128, 1)
out_proj(const __bf16* __restrict__ ctx, const __bf16* __restrict__ Wpb,
         const float* __restrict__ bp, float* __restrict__ out) {
  const int lane = threadIdx.x;
  const int ln   = lane & 15;
  const int half = lane >> 4;
  const int kb   = half * 8;
  const int mt   = blockIdx.x;
  const int n64  = blockIdx.y * blockDim.y + threadIdx.y;

  const __bf16* cr0 = ctx + (size_t)(mt * 32 + ln) * EMBED;
  const __bf16* cr1 = cr0 + (size_t)16 * EMBED;
  const __bf16* w0  = Wpb + (size_t)(n64 * 64 + ln) * EMBED;

  v8f acc[2][4] = {};
  gemm_loop(acc, cr0, cr1, w0, kb);

#pragma unroll
  for (int i = 0; i < 2; ++i) {
#pragma unroll
    for (int j = 0; j < 4; ++j) {
      const int ncol = n64 * 64 + j * 16 + ln;
      const float bval = bp[ncol];
#pragma unroll
      for (int r = 0; r < 8; ++r) {
        int mr = mt * 32 + i * 16 + r + 8 * half;
        if (mr >= MTOT) break;
        out[(size_t)mr * EMBED + ncol] = acc[i][j][r] + bval;
      }
    }
  }
}

extern "C" void kernel_launch(void* const* d_in, const int* in_sizes, int n_in,
                              void* d_out, int out_size, void* d_ws, size_t ws_size,
                              hipStream_t stream) {
  (void)in_sizes; (void)n_in; (void)out_size; (void)ws_size;
  const float* x  = (const float*)d_in[0];
  const float* Wk = (const float*)d_in[1];
  const float* bk = (const float*)d_in[2];
  const float* Wq = (const float*)d_in[3];
  const float* bq = (const float*)d_in[4];
  const float* Wv = (const float*)d_in[5];
  const float* bv = (const float*)d_in[6];
  const float* Wp = (const float*)d_in[7];
  const float* bp = (const float*)d_in[8];
  float* out = (float*)d_out;

  char* ws = (char*)d_ws;
  size_t off = 0;
  auto salloc = [&](size_t bytes) -> void* {
    void* p = ws + off;
    off += (bytes + 255) & ~(size_t)255;
    return p;
  };
  __bf16* xb   = (__bf16*)salloc((size_t)MPAD * EMBED * 2);
  __bf16* Wqb  = (__bf16*)salloc((size_t)EMBED * EMBED * 2);
  __bf16* Wkb  = (__bf16*)salloc((size_t)EMBED * EMBED * 2);
  __bf16* Wvb  = (__bf16*)salloc((size_t)EMBED * EMBED * 2);
  __bf16* Wpb  = (__bf16*)salloc((size_t)EMBED * EMBED * 2);
  __bf16* qbuf = (__bf16*)salloc((size_t)BATCH * NHEAD * SPAD * HD * 2);
  __bf16* kbuf = (__bf16*)salloc((size_t)BATCH * NHEAD * SPAD * HD * 2);
  __bf16* vtb  = (__bf16*)salloc((size_t)BATCH * NHEAD * HD * SPAD * 2);
  __bf16* ctx  = (__bf16*)salloc((size_t)MPAD * EMBED * 2);

  int n = MTOT * EMBED;
  cvt_bf16<<<(n + 255) / 256, 256, 0, stream>>>(x, xb, n);
  n = EMBED * EMBED;
  cvt_bf16<<<(n + 255) / 256, 256, 0, stream>>>(Wq, Wqb, n);
  cvt_bf16<<<(n + 255) / 256, 256, 0, stream>>>(Wk, Wkb, n);
  cvt_bf16<<<(n + 255) / 256, 256, 0, stream>>>(Wv, Wvb, n);
  cvt_bf16<<<(n + 255) / 256, 256, 0, stream>>>(Wp, Wpb, n);

  dim3 blk(32, 4);
  dim3 gq((MPAD / 32), (EMBED / 64) / 4, 3);           // 257 x 3 x 3
  qkv_gemm<<<gq, blk, 0, stream>>>(xb, Wqb, Wkb, Wvb, bq, bk, bv, qbuf, kbuf, vtb);

  int qtiles = (SEQ + 15) / 16;                        // 65
  dim3 ga((qtiles + 3) / 4, NHEAD, BATCH);             // 17 x 12 x 8
  attn_kernel<<<ga, blk, 0, stream>>>(qbuf, kbuf, vtb, ctx);

  dim3 gp((MPAD / 32), (EMBED / 64) / 4);              // 257 x 3
  out_proj<<<gp, blk, 0, stream>>>(ctx, Wpb, bp, out);
}
